// MultiHeadAttention_82274393522832
// MI455X (gfx1250) — matrix operand order repaired and checked by reading, not compiled, not run on hardware
//
#include <hip/hip_runtime.h>

// B=2, S=2048, D=1024, H=16, HD=64 causal MHA for MI455X (gfx1250, wave32).
#define B_  2
#define S_  2048
#define D_  1024
#define H_  16
#define HD_ 64

typedef __attribute__((ext_vector_type(16))) __bf16          v16bf;
typedef __attribute__((ext_vector_type(8)))  __bf16          v8bf;
typedef __attribute__((ext_vector_type(8)))  float           v8f;
typedef __attribute__((ext_vector_type(8)))  unsigned short  u16x8;
typedef __attribute__((ext_vector_type(16))) unsigned short  u16x16;

// fp32 -> bf16 via native conversion (backend picks v_cvt path, RNE)
static __device__ __forceinline__ unsigned short f2bf(float f) {
    __bf16 h = (__bf16)f;
    return __builtin_bit_cast(unsigned short, h);
}

static __device__ __forceinline__ v8f wmma_bf16(v16bf a, v16bf b, v8f c) {
    // D = A(16x32 bf16) * B(32x16 bf16) + C(16x16 f32)
    return __builtin_amdgcn_wmma_f32_16x16x32_bf16(false, a, false, b, (short)0, c, false, false);
}

static __device__ __forceinline__ v16bf mk16(u16x8 lo, u16x8 hi) {
    u16x16 v = __builtin_shufflevector(lo, hi, 0,1,2,3,4,5,6,7,8,9,10,11,12,13,14,15);
    return __builtin_bit_cast(v16bf, v);
}

// A operand (16x32, MxK): lane l<16 -> row l, K {kb+0..15};
// lane l>=16 -> row l-16, K {kb+16..31}. Two 16B loads per lane.
static __device__ __forceinline__ v16bf load_Abf(const unsigned short* base, int ld, int kb, int lane) {
    int n = lane & 15, half = lane >> 4;
    const unsigned short* p = base + n * ld + kb + half * 16;
    u16x8 lo = *(const u16x8*)p;
    u16x8 hi = *(const u16x8*)(p + 8);
    return mk16(lo, hi);
}

// Same A layout, but source is fp32: load 2x v8f, packed-convert to bf16.
static __device__ __forceinline__ v16bf load_Af32(const float* base, int ld, int kb, int lane) {
    int n = lane & 15, half = lane >> 4;
    const float* p = base + n * ld + kb + half * 16;
    v8bf lo = __builtin_convertvector(*(const v8f*)p,        v8bf);
    v8bf hi = __builtin_convertvector(*(const v8f*)(p + 8), v8bf);
    return __builtin_shufflevector(lo, hi, 0,1,2,3,4,5,6,7,8,9,10,11,12,13,14,15);
}

// B operand (32x16, KxN) supplied as rows of B^T: lane l -> column n=l&15,
// K = kb + 16*(l>>4) .. +15 consecutive. One 32B contiguous load per lane.
static __device__ __forceinline__ v16bf load_Bbf(const unsigned short* bT, int ld, int kb, int lane) {
    int n = lane & 15, half = lane >> 4;
    const u16x16* p = (const u16x16*)(bT + n * ld + kb + half * 16);
    return __builtin_bit_cast(v16bf, *p);
}

// ---------------------------------------------------------------- kernel 1
// Transposing bf16 convert: in fp32 [Hn][Dn][En] -> out bf16 [Hn][En][Dn]
__global__ void wtrans_kernel(const float* __restrict__ in, unsigned short* __restrict__ out,
                              int Hn, int Dn, int En) {
    int idx = blockIdx.x * blockDim.x + threadIdx.x;
    int total = Hn * Dn * En;
    if (idx >= total) return;
    int e = idx % En;
    int t = idx / En;
    int d = t % Dn;
    int h = t / Dn;
    out[(h * En + e) * Dn + d] = f2bf(in[idx]);
}

// ---------------------------------------------------------------- kernel 2
// Per-head projections. z=0: Q = (queries@Wq)*0.125 -> Qp[bh][s][e]  (1/sqrt(HD) folded in)
//                       z=1: K = keys@Wk            -> Kp[bh][s][e]
//                       z=2: V = values@Wv          -> Vt[bh][e][s]  (transposed!)
__global__ void __launch_bounds__(128)
proj_kernel(const float* __restrict__ qin, const float* __restrict__ kin,
            const float* __restrict__ vin,
            const unsigned short* __restrict__ WqT, const unsigned short* __restrict__ WkT,
            const unsigned short* __restrict__ WvT,
            unsigned short* __restrict__ Qp, unsigned short* __restrict__ Kp,
            unsigned short* __restrict__ Vt) {
    int lane  = threadIdx.x & 31;
    int wv    = threadIdx.x >> 5;
    int sTile = blockIdx.x * 4 + wv;         // 0..127 (16 rows each)
    int bh    = blockIdx.y;                  // 0..31
    int b     = bh >> 4, h = bh & 15;
    int z     = blockIdx.z;                  // matrix select

    const float* in = (z == 0 ? qin : (z == 1 ? kin : vin)) + (b * S_ + sTile * 16) * D_;
    const unsigned short* WT = (z == 0 ? WqT : (z == 1 ? WkT : WvT)) + h * HD_ * D_;

    v8f c[4] = {};
    for (int kb = 0; kb < D_; kb += 32) {
        v16bf a = load_Af32(in, D_, kb, lane);
#pragma unroll
        for (int j = 0; j < 4; ++j)
            c[j] = wmma_bf16(a, load_Bbf(WT + j * 16 * D_, D_, kb, lane), c[j]);
    }

    int n = lane & 15, half = lane >> 4;
    float scl = (z == 0) ? 0.125f : 1.0f;    // fold 1/sqrt(64) into Q
    if (z < 2) {
        unsigned short* outp = (z == 0 ? Qp : Kp) + (bh * S_ + sTile * 16) * HD_;
#pragma unroll
        for (int g = 0; g < 8; ++g) {
            int r = g + 8 * half;
#pragma unroll
            for (int j = 0; j < 4; ++j)
                outp[r * HD_ + j * 16 + n] = f2bf(c[j][g] * scl);
        }
    } else {
        unsigned short* outp = Vt + bh * HD_ * S_;   // [e][s]
#pragma unroll
        for (int g = 0; g < 8; ++g) {
            int r = g + 8 * half;
#pragma unroll
            for (int j = 0; j < 4; ++j)
                outp[(j * 16 + n) * S_ + sTile * 16 + r] = f2bf(c[j][g]);
        }
    }
}

// ---------------------------------------------------------------- kernel 3
// Flash attention, one wave per 16-row q-tile, 32 keys per step.
__global__ void __launch_bounds__(128)
attn_kernel(const unsigned short* __restrict__ Qp, const unsigned short* __restrict__ Kp,
            const unsigned short* __restrict__ Vt, unsigned short* __restrict__ X) {
    __shared__ __align__(16) unsigned short Pb[4][16 * 32];   // per-wave P staging

    int lane = threadIdx.x & 31;
    int wv   = threadIdx.x >> 5;
    int qt   = blockIdx.x * 4 + wv;          // 0..127
    int bh   = blockIdx.y;                   // 0..31
    int b    = bh >> 4, h = bh & 15;
    int q0   = qt * 16;
    int n    = lane & 15, half = lane >> 4;

    const unsigned short* Qb = Qp + (bh * S_ + q0) * HD_;
    const unsigned short* Kb = Kp + bh * S_ * HD_;
    const unsigned short* Vb = Vt + bh * HD_ * S_;

    // Q tile (16x64, pre-scaled by 1/sqrt(HD)) kept in registers as two A operands.
    v16bf qA0 = load_Abf(Qb, HD_, 0, lane);
    v16bf qA1 = load_Abf(Qb, HD_, 32, lane);

    v8f o0 = {}, o1 = {}, o2 = {}, o3 = {};
    float m[8], l[8];
#pragma unroll
    for (int g = 0; g < 8; ++g) { m[g] = -1e30f; l[g] = 0.0f; }

    for (int kb0 = 0; kb0 <= q0 + 15; kb0 += 32) {
        __builtin_prefetch(Kb + (kb0 + 32) * HD_, 0, 0);
        // scores: two 16-key sub-tiles over feature K=64 (2 chunks of 32)
        v8f s0 = {}, s1 = {};
        s0 = wmma_bf16(qA0, load_Bbf(Kb + kb0 * HD_, HD_, 0,  lane), s0);
        s0 = wmma_bf16(qA1, load_Bbf(Kb + kb0 * HD_, HD_, 32, lane), s0);
        s1 = wmma_bf16(qA0, load_Bbf(Kb + (kb0 + 16) * HD_, HD_, 0,  lane), s1);
        s1 = wmma_bf16(qA1, load_Bbf(Kb + (kb0 + 16) * HD_, HD_, 32, lane), s1);

        bool needmask = (kb0 + 31 > q0);
#pragma unroll
        for (int g = 0; g < 8; ++g) {
            int row = g + 8 * half;                     // 0..15 within tile
            float a0 = s0[g];
            float a1 = s1[g];
            if (needmask) {
                if (kb0 + n      > q0 + row) a0 = -1e30f;
                if (kb0 + 16 + n > q0 + row) a1 = -1e30f;
            }
            // row max across the 16 lanes holding this row's columns
            float mx = fmaxf(a0, a1);
#pragma unroll
            for (int mm = 1; mm < 16; mm <<= 1) mx = fmaxf(mx, __shfl_xor(mx, mm));
            float mnew = fmaxf(m[g], mx);
            float p0 = __expf(a0 - mnew);
            float p1 = __expf(a1 - mnew);
            float sum = p0 + p1;
#pragma unroll
            for (int mm = 1; mm < 16; mm <<= 1) sum += __shfl_xor(sum, mm);
            float sc = __expf(m[g] - mnew);
            l[g] = l[g] * sc + sum;
            m[g] = mnew;
            o0[g] *= sc; o1[g] *= sc; o2[g] *= sc; o3[g] *= sc;
            // stage P (C-layout -> row-major bf16 in LDS)
            Pb[wv][row * 32 + n]      = f2bf(p0);
            Pb[wv][row * 32 + 16 + n] = f2bf(p1);
        }
        asm volatile("s_wait_dscnt 0" ::: "memory");

        // reload P in A-operand layout from LDS (ds_load_b128 x2)
        u16x8 plo = *(const u16x8*)&Pb[wv][n * 32 + half * 16];
        u16x8 phi = *(const u16x8*)&Pb[wv][n * 32 + half * 16 + 8];
        v16bf aP = mk16(plo, phi);

        // O(16x64) += P(16x32) @ V(32x16) x4, V^T rows are contiguous
        o0 = wmma_bf16(aP, load_Bbf(Vb + 0  * S_, S_, kb0, lane), o0);
        o1 = wmma_bf16(aP, load_Bbf(Vb + 16 * S_, S_, kb0, lane), o1);
        o2 = wmma_bf16(aP, load_Bbf(Vb + 32 * S_, S_, kb0, lane), o2);
        o3 = wmma_bf16(aP, load_Bbf(Vb + 48 * S_, S_, kb0, lane), o3);
    }

    // finalize: divide by l, write concat buffer X[b][s][h*64+e] (bf16)
    unsigned short* xp = X + (b * S_ + q0) * (H_ * HD_) + h * HD_;
#pragma unroll
    for (int g = 0; g < 8; ++g) {
        float inv = 1.0f / l[g];
        int r = g + 8 * half;
        unsigned short* rowp = xp + r * (H_ * HD_);
        rowp[0 * 16 + n] = f2bf(o0[g] * inv);
        rowp[1 * 16 + n] = f2bf(o1[g] * inv);
        rowp[2 * 16 + n] = f2bf(o2[g] * inv);
        rowp[3 * 16 + n] = f2bf(o3[g] * inv);
    }
}

// ---------------------------------------------------------------- kernel 4
// out(4096x1024 f32) = X(bf16) @ Wo ; WoT holds Wo^T rows [out][in]
__global__ void __launch_bounds__(128)
outproj_kernel(const unsigned short* __restrict__ X, const unsigned short* __restrict__ WoT,
               float* __restrict__ out) {
    int lane = threadIdx.x & 31;
    int wv   = threadIdx.x >> 5;
    int rt   = blockIdx.x * 4 + wv;          // 0..255 row tiles
    int ct   = blockIdx.y;                   // 0..15 col tiles of 64

    const unsigned short* A = X + rt * 16 * D_;
    v8f c[4] = {};
    for (int kb = 0; kb < D_; kb += 32) {
        v16bf a = load_Abf(A, D_, kb, lane);
#pragma unroll
        for (int j = 0; j < 4; ++j)
            c[j] = wmma_bf16(a, load_Bbf(WoT + (ct * 64 + j * 16) * D_, D_, kb, lane), c[j]);
    }
    int n = lane & 15, half = lane >> 4;
    float* op = out + rt * 16 * D_ + ct * 64;
#pragma unroll
    for (int g = 0; g < 8; ++g) {
        int r = g + 8 * half;
#pragma unroll
        for (int j = 0; j < 4; ++j)
            op[r * D_ + j * 16 + n] = c[j][g];
    }
}

// ---------------------------------------------------------------- launcher
extern "C" void kernel_launch(void* const* d_in, const int* in_sizes, int n_in,
                              void* d_out, int out_size, void* d_ws, size_t ws_size,
                              hipStream_t stream) {
    (void)in_sizes; (void)n_in; (void)out_size; (void)ws_size;
    const float* keys    = (const float*)d_in[0];
    const float* queries = (const float*)d_in[1];
    const float* values  = (const float*)d_in[2];
    const float* Wq      = (const float*)d_in[3];
    const float* Wk      = (const float*)d_in[4];
    const float* Wv      = (const float*)d_in[5];
    const float* Wo      = (const float*)d_in[6];

    // Workspace layout (bf16 elements), total 40 MiB.
    unsigned short* ws = (unsigned short*)d_ws;
    const size_t W1 = (size_t)H_ * HD_ * D_;        // 1,048,576
    const size_t P1 = (size_t)B_ * H_ * S_ * HD_;   // 4,194,304
    unsigned short* WqT = ws;
    unsigned short* WkT = WqT + W1;
    unsigned short* WvT = WkT + W1;
    unsigned short* WoT = WvT + W1;
    unsigned short* Qp  = WoT + (size_t)D_ * D_;
    unsigned short* Kp  = Qp + P1;
    unsigned short* Vt  = Kp + P1;
    unsigned short* Xc  = Vt + P1;

    const int ctb = 256;
    int wtot = H_ * D_ * HD_;
    wtrans_kernel<<<(wtot + ctb - 1) / ctb, ctb, 0, stream>>>(Wq, WqT, H_, D_, HD_);
    wtrans_kernel<<<(wtot + ctb - 1) / ctb, ctb, 0, stream>>>(Wk, WkT, H_, D_, HD_);
    wtrans_kernel<<<(wtot + ctb - 1) / ctb, ctb, 0, stream>>>(Wv, WvT, H_, D_, HD_);
    wtrans_kernel<<<(D_ * D_ + ctb - 1) / ctb, ctb, 0, stream>>>(Wo, WoT, 1, H_ * HD_, D_);

    proj_kernel<<<dim3(S_ / 64, B_ * H_, 3), 128, 0, stream>>>(
        queries, keys, values, WqT, WkT, WvT, Qp, Kp, Vt);

    attn_kernel<<<dim3(S_ / 64, B_ * H_), 128, 0, stream>>>(Qp, Kp, Vt, Xc);

    outproj_kernel<<<dim3(64, 16), 128, 0, stream>>>(Xc, WoT, (float*)d_out);
}
